// GraphTransformerEdgePredictor_59923383714101
// MI455X (gfx1250) — matrix operand order repaired and checked by reading, not compiled, not run on hardware
//
#include <hip/hip_runtime.h>
#include <math.h>
#include <stdint.h>

// Problem constants (match reference).
#define NN   4096        // nodes
#define DD   512         // hidden
#define HH   8           // heads
#define DHD  64          // head dim
#define EE   262144      // edges
#define DFFN 2048        // 4*D
#define D2D  1024        // 2*D
#define CEC  8192        // edge chunk size

typedef __attribute__((ext_vector_type(16))) _Float16 v16h;
typedef __attribute__((ext_vector_type(8)))  _Float16 v8h;
typedef __attribute__((ext_vector_type(8)))  float    v8f;

#define F_RELU 1

// ---------------------------------------------------------------------------
// Async global->LDS 16-byte copy (CDNA5 ASYNCcnt path).
// LDS address = low 32 bits of the flat pointer (flat->LDS aperture mapping).
// ---------------------------------------------------------------------------
__device__ __forceinline__ void gt_async_b128(_Float16* lds_dst, const _Float16* g_src)
{
    uint32_t l = (uint32_t)(uintptr_t)lds_dst;
    asm volatile("global_load_async_to_lds_b128 %0, %1, off"
                 :: "v"(l), "v"((unsigned long long)(uintptr_t)g_src)
                 : "memory");
}

// ---------------------------------------------------------------------------
// WMMA GEMM: C[M,Nc] (f32, opt) / Ch (f16, opt) = act(A[M,K] * B^T) (+bias)(+res)
// A: f16 [M,K] with leading dim lda; B: f16 [Nc,K] with leading dim ldb
// (i.e. B is always pre-laid-out n-major, K contiguous).
// Requires M%128==0, Nc%64==0, K%32==0 (true for all uses here).
// Block: 256 threads (8 waves). Block tile 128x64, wave tile 32x32, BK=32.
// Double-buffered async global->LDS staging.
// ---------------------------------------------------------------------------
__global__ __launch_bounds__(256)
void gt_gemm_wmma(const _Float16* __restrict__ A, int lda,
                  const _Float16* __restrict__ B, int ldb,
                  float* __restrict__ C, _Float16* __restrict__ Ch, int ldc,
                  const float* __restrict__ bias,
                  const float* __restrict__ res,
                  int M, int Nc, int K, int flags)
{
    __shared__ __align__(64) _Float16 As[2][128 * 32];   // [row][k]
    __shared__ __align__(64) _Float16 Bs[2][64 * 32];    // [col][k]

    const int tid  = threadIdx.x;
    const int lane = tid & 31;
    const int wid  = tid >> 5;
    const int wm   = wid & 3;          // wave row tile (32 rows each)
    const int wn   = wid >> 2;         // wave col tile (32 cols each)
    const int bm   = blockIdx.y * 128;
    const int bn   = blockIdx.x * 64;
    const int hi   = lane >> 4;        // lane half
    const int l15  = lane & 15;

    // Issue one K-tile of async copies: A 128x32 (512 x 16B chunks, 2/thread),
    // B 64x32 (256 chunks, 1/thread). 3 async instructions per wave per tile.
    auto issue_tile = [&](int buf, int kk) {
        _Float16* as = &As[buf][0];
        _Float16* bs = &Bs[buf][0];
        #pragma unroll
        for (int i = 0; i < 2; ++i) {
            int c  = tid + (i << 8);
            int r  = c >> 2, k8 = (c & 3) << 3;
            gt_async_b128(as + r * 32 + k8,
                          A + (size_t)(bm + r) * lda + kk + k8);
        }
        int r  = tid >> 2, k8 = (tid & 3) << 3;
        gt_async_b128(bs + r * 32 + k8,
                      B + (size_t)(bn + r) * ldb + kk + k8);
    };

    v8f acc[2][2] = {};
    const int nsteps = K >> 5;

    issue_tile(0, 0);
    for (int step = 0; step < nsteps; ++step) {
        const int cur = step & 1;
        if (step + 1 < nsteps) {
            issue_tile(cur ^ 1, (step + 1) << 5);   // prefetch next tile
            asm volatile("s_wait_asynccnt 3" ::: "memory");  // cur tile done
        } else {
            asm volatile("s_wait_asynccnt 0" ::: "memory");
        }
        __syncthreads();

        const _Float16* as = &As[cur][0];
        const _Float16* bs = &Bs[cur][0];
        const int koff = hi * 8;

        // A fragments (lane<16 -> K 0..15, lane>=16 -> K 16..31)
        union { v16h v; v8h h[2]; } a0, a1;
        const int ar0 = wm * 32 + l15;
        a0.v = *(const v16h*)(as + ar0 * 32 + hi * 16);
        a1.v = *(const v16h*)(as + (ar0 + 16) * 32 + hi * 16);
        // B fragments (lane<16 -> K 0..15, lane>=16 -> K 16..31)
        const int bc0 = wn * 32 + l15;
        v16h b0 = *(const v16h*)(bs + bc0 * 32 + hi * 16);
        v16h b1 = *(const v16h*)(bs + (bc0 + 16) * 32 + hi * 16);

        acc[0][0] = __builtin_amdgcn_wmma_f32_16x16x32_f16(false, a0.v, false, b0,
                                                           (short)0, acc[0][0], false, false);
        acc[0][1] = __builtin_amdgcn_wmma_f32_16x16x32_f16(false, a0.v, false, b1,
                                                           (short)0, acc[0][1], false, false);
        acc[1][0] = __builtin_amdgcn_wmma_f32_16x16x32_f16(false, a1.v, false, b0,
                                                           (short)0, acc[1][0], false, false);
        acc[1][1] = __builtin_amdgcn_wmma_f32_16x16x32_f16(false, a1.v, false, b1,
                                                           (short)0, acc[1][1], false, false);
        __syncthreads();   // readers done before next issue overwrites this buffer
    }

    // Epilogue. C-frag layout: VGPR j -> row hi*8+j, col l15.
    #pragma unroll
    for (int fi = 0; fi < 2; ++fi) {
        #pragma unroll
        for (int fj = 0; fj < 2; ++fj) {
            const int ro = bm + wm * 32 + fi * 16 + hi * 8;
            const int co = bn + wn * 32 + fj * 16 + l15;
            float bv = bias ? bias[co] : 0.0f;
            #pragma unroll
            for (int j = 0; j < 8; ++j) {
                float v = acc[fi][fj][j] + bv;
                if (flags & F_RELU) v = fmaxf(v, 0.0f);
                if (res) v += res[(size_t)(ro + j) * ldc + co];
                if (C)  C[(size_t)(ro + j) * ldc + co] = v;
                if (Ch) Ch[(size_t)(ro + j) * ldc + co] = (_Float16)v;
            }
        }
    }
}

// ---------------------------------------------------------------------------
// f32 -> f16 conversion helpers.
// ---------------------------------------------------------------------------
__global__ void gt_cvt_f16(const float* __restrict__ src, int ld,
                           _Float16* __restrict__ dst, int rows, int cols)
{
    int idx = blockIdx.x * blockDim.x + threadIdx.x;
    if (idx >= rows * cols) return;
    int r = idx / cols, c = idx - r * cols;
    dst[idx] = (_Float16)src[(size_t)r * ld + c];
}
// dst[n*K + k] = src[k*ld + n]  (f16 src; builds [Nc][K] from [K,Nc] slice)
__global__ void gt_cvt_t_f16(const _Float16* __restrict__ src, int ld,
                             _Float16* __restrict__ dst, int K, int Nc)
{
    int idx = blockIdx.x * blockDim.x + threadIdx.x;
    if (idx >= K * Nc) return;
    int n = idx / K, k = idx - n * K;
    dst[idx] = src[(size_t)k * ld + n];
}

// ---------------------------------------------------------------------------
// Block-wide reduction (256 threads, wave32).
// ---------------------------------------------------------------------------
__device__ __forceinline__ float gt_blk_reduce(float v, bool do_max)
{
    __shared__ float red[8];
    const int lane = threadIdx.x & 31;
    const int wid  = threadIdx.x >> 5;
    #pragma unroll
    for (int o = 16; o; o >>= 1) {
        float ov = __shfl_xor(v, o, 32);
        v = do_max ? fmaxf(v, ov) : (v + ov);
    }
    __syncthreads();
    if (lane == 0) red[wid] = v;
    __syncthreads();
    float r = red[0];
    #pragma unroll
    for (int j = 1; j < 8; ++j) r = do_max ? fmaxf(r, red[j]) : (r + red[j]);
    return r;
}

// ---------------------------------------------------------------------------
// Row LayerNorm: y = [res +] act((in-mu)*rsqrt(var+eps)*g + b)
// Writes f32 (out) and/or f16 (outh). One block per row; D in {256,512,1024}.
// ---------------------------------------------------------------------------
__global__ __launch_bounds__(256)
void gt_row_ln(const float* __restrict__ in, const float* __restrict__ res,
               float* __restrict__ out, _Float16* __restrict__ outh,
               const float* __restrict__ g, const float* __restrict__ b,
               int D, int flags)
{
    const int row = blockIdx.x;
    const int tid = threadIdx.x;
    const int per = D >> 8;                 // 1, 2 or 4
    const float* p = in + (size_t)row * D;
    float v[4];
    float s = 0.f, sq = 0.f;
    for (int i = 0; i < per; ++i) {
        float x = p[tid + (i << 8)];
        v[i] = x; s += x; sq += x * x;
    }
    s  = gt_blk_reduce(s,  false);
    sq = gt_blk_reduce(sq, false);
    float mu  = s / (float)D;
    float var = sq / (float)D - mu * mu;
    float inv = rsqrtf(var + 1e-5f);
    for (int i = 0; i < per; ++i) {
        int c = tid + (i << 8);
        float y = (v[i] - mu) * inv * g[c] + b[c];
        if (flags & 1) y = fmaxf(y, 0.f);
        if (flags & 2) y += res[(size_t)row * D + c];
        if (out)  out[(size_t)row * D + c] = y;
        if (outh) outh[(size_t)row * D + c] = (_Float16)y;
    }
}

// Row softmax over f16 scores (in-place), ncols = 4096.
__global__ __launch_bounds__(256)
void gt_softmax_rows(_Float16* __restrict__ S, int ncols, float scale)
{
    const int row = blockIdx.x;
    const int tid = threadIdx.x;
    _Float16* p = S + (size_t)row * ncols;
    float v[16];
    float mx = -INFINITY;
    #pragma unroll
    for (int i = 0; i < 16; ++i) {
        v[i] = (float)p[tid + (i << 8)] * scale;
        mx = fmaxf(mx, v[i]);
    }
    mx = gt_blk_reduce(mx, true);
    float s = 0.f;
    #pragma unroll
    for (int i = 0; i < 16; ++i) { v[i] = __expf(v[i] - mx); s += v[i]; }
    s = gt_blk_reduce(s, false);
    float inv = 1.f / s;
    #pragma unroll
    for (int i = 0; i < 16; ++i) p[tid + (i << 8)] = (_Float16)(v[i] * inv);
}

// ---------------------------------------------------------------------------
// GCN helpers
// ---------------------------------------------------------------------------
__global__ void gt_fill(float* p, float v, int n)
{
    int i = blockIdx.x * blockDim.x + threadIdx.x;
    if (i < n) p[i] = v;
}
__global__ void gt_deg_accum(float* deg, const int* __restrict__ dst, int E)
{
    int e = blockIdx.x * blockDim.x + threadIdx.x;
    if (e < E) atomicAdd(&deg[dst[e]], 1.0f);
}
__global__ void gt_rsqrt_ip(float* p, int n)
{
    int i = blockIdx.x * blockDim.x + threadIdx.x;
    if (i < n) p[i] = rsqrtf(p[i]);
}
__global__ void gt_gcn_self(const float* __restrict__ xw,
                            const float* __restrict__ dinv,
                            const float* __restrict__ bias,
                            float* __restrict__ out, int n)
{
    int idx = blockIdx.x * blockDim.x + threadIdx.x;
    if (idx >= n) return;
    int r = idx >> 9, c = idx & 511;
    float di = dinv[r];
    out[idx] = xw[idx] * di * di + bias[c];
}
__global__ void gt_gcn_edge(const float* __restrict__ xw,
                            const int* __restrict__ src,
                            const int* __restrict__ dst,
                            const float* __restrict__ dinv,
                            float* __restrict__ out, int total)
{
    int idx = blockIdx.x * blockDim.x + threadIdx.x;
    if (idx >= total) return;
    int e = idx >> 9, c = idx & 511;
    int s = src[e], d = dst[e];
    atomicAdd(&out[(size_t)d * 512 + c], xw[(size_t)s * 512 + c] * dinv[s] * dinv[d]);
}
__global__ void gt_pe_add(float* __restrict__ x, int n)
{
    int idx = blockIdx.x * blockDim.x + threadIdx.x;
    if (idx >= n) return;
    int nn = idx >> 9, c = idx & 511;
    float di  = __expf((float)(c & ~1) * (-logf(10000.f) / 512.f));
    float ang = (float)nn * di;
    x[idx] += (c & 1) ? cosf(ang) : sinf(ang);
}

// ---------------------------------------------------------------------------
// Edge stage
// ---------------------------------------------------------------------------
// One block per edge: LN(concat(x[src],x[dst])) -> f16 row (GEMM operand).
__global__ __launch_bounds__(256)
void gt_edge_gather_ln(const float* __restrict__ x,
                       const int* __restrict__ src, const int* __restrict__ dst,
                       const float* __restrict__ g, const float* __restrict__ b,
                       _Float16* __restrict__ out)
{
    const int e  = blockIdx.x;
    const int tid = threadIdx.x;
    const int s = src[e], d = dst[e];
    float v[4];
    float su = 0.f, sq = 0.f;
    #pragma unroll
    for (int i = 0; i < 4; ++i) {
        int c = tid + (i << 8);
        float xv = (c < 512) ? x[(size_t)s * 512 + c] : x[(size_t)d * 512 + (c - 512)];
        v[i] = xv; su += xv; sq += xv * xv;
    }
    su = gt_blk_reduce(su, false);
    sq = gt_blk_reduce(sq, false);
    float mu  = su * (1.f / 1024.f);
    float var = sq * (1.f / 1024.f) - mu * mu;
    float inv = rsqrtf(var + 1e-5f);
    #pragma unroll
    for (int i = 0; i < 4; ++i) {
        int c = tid + (i << 8);
        out[(size_t)e * 1024 + c] = (_Float16)((v[i] - mu) * inv * g[c] + b[c]);
    }
}
__global__ void gt_edge_out(const float* __restrict__ h2,
                            const float* __restrict__ w3,
                            const float* __restrict__ b3,
                            float* __restrict__ out, int n)
{
    int e = blockIdx.x * blockDim.x + threadIdx.x;
    if (e >= n) return;
    const float* h = h2 + (size_t)e * 256;
    float acc = 0.f;
    #pragma unroll 4
    for (int j = 0; j < 256; ++j) acc += h[j] * w3[j];
    acc += b3[0];
    out[e] = 1.f / (1.f + __expf(-acc));
}

// ---------------------------------------------------------------------------
// Host orchestration
// ---------------------------------------------------------------------------
static inline void gemm16(hipStream_t st, const _Float16* A, int lda,
                          const _Float16* B, int ldb,
                          float* C, _Float16* Ch, int ldc,
                          const float* bias, const float* res,
                          int M, int Nc, int K, int flags)
{
    dim3 g(Nc / 64, M / 128);
    gt_gemm_wmma<<<g, 256, 0, st>>>(A, lda, B, ldb, C, Ch, ldc, bias, res,
                                    M, Nc, K, flags);
}

extern "C" void kernel_launch(void* const* d_in, const int* in_sizes, int n_in,
                              void* d_out, int out_size, void* d_ws, size_t ws_size,
                              hipStream_t stream)
{
    (void)in_sizes; (void)n_in; (void)out_size;
    int ii = 0;
    const int*   edge_index = (const int*)d_in[ii++];   // [2,E]
    const float* node_emb   = (const float*)d_in[ii++]; // [N,D]
    struct Gcn { const float *W, *b, *g, *be; } gcn[2];
    for (int l = 0; l < 2; ++l) {
        gcn[l].W  = (const float*)d_in[ii++];
        gcn[l].b  = (const float*)d_in[ii++];
        gcn[l].g  = (const float*)d_in[ii++];
        gcn[l].be = (const float*)d_in[ii++];
    }
    struct Tr { const float *Wqkv,*bqkv,*Wo,*bo,*g1,*b1n,*g2,*b2n,*W1,*bf1,*W2,*bf2; } tr[2];
    for (int l = 0; l < 2; ++l) {
        tr[l].Wqkv = (const float*)d_in[ii++]; tr[l].bqkv = (const float*)d_in[ii++];
        tr[l].Wo   = (const float*)d_in[ii++]; tr[l].bo   = (const float*)d_in[ii++];
        tr[l].g1   = (const float*)d_in[ii++]; tr[l].b1n  = (const float*)d_in[ii++];
        tr[l].g2   = (const float*)d_in[ii++]; tr[l].b2n  = (const float*)d_in[ii++];
        tr[l].W1   = (const float*)d_in[ii++]; tr[l].bf1  = (const float*)d_in[ii++];
        tr[l].W2   = (const float*)d_in[ii++]; tr[l].bf2  = (const float*)d_in[ii++];
    }
    const float* pre_g = (const float*)d_in[ii++];
    const float* pre_b = (const float*)d_in[ii++];
    const float *mW1 = (const float*)d_in[ii++], *mb1 = (const float*)d_in[ii++];
    const float *mg1 = (const float*)d_in[ii++], *mbe1 = (const float*)d_in[ii++];
    const float *mW2 = (const float*)d_in[ii++], *mb2 = (const float*)d_in[ii++];
    const float *mg2 = (const float*)d_in[ii++], *mbe2 = (const float*)d_in[ii++];
    const float *mW3 = (const float*)d_in[ii++], *mb3 = (const float*)d_in[ii++];

    const int* src = edge_index;
    const int* dst = edge_index + EE;
    float* outp = (float*)d_out;

    // ---- Workspace layout: f32 region then f16 region ----
    float* ws = (float*)d_ws;
    size_t off = 0;
    float* xbuf = ws + off; off += (size_t)NN * DD;      // node features (f32)
    float* t1   = ws + off; off += (size_t)NN * DD;      // GCN xw
    float* newx = ws + off; off += (size_t)NN * DD;      // GCN aggregate
    float* obuf = ws + off; off += (size_t)NN * DD;      // attention output
    float* eh1  = ws + off; off += (size_t)CEC * DD;     // edge h1
    float* eh2  = ws + off; off += (size_t)CEC * (DD/2); // edge h2
    float* degb = ws + off; off += (size_t)NN;           // deg^-1/2

    _Float16* hb = (_Float16*)(ws + off);
    size_t ho = 0;
    _Float16* xh   = hb + ho; ho += (size_t)NN * DD;        // x (f16)
    _Float16* lnh  = hb + ho; ho += (size_t)NN * DD;        // LN output
    _Float16* qkvh = hb + ho; ho += (size_t)NN * 3 * DD;    // QKV (f16)
    _Float16* scbh = hb + ho; ho += (size_t)NN * NN;        // per-head scores
    _Float16* vth  = hb + ho; ho += (size_t)DHD * NN;       // V^T per head
    _Float16* oh   = hb + ho; ho += (size_t)NN * DD;        // attn out (f16)
    _Float16* t1h  = hb + ho; ho += (size_t)NN * DFFN;      // FFN hidden (f16)
    _Float16* whb  = hb + ho; ho += (size_t)DFFN * DD;      // weight scratch
    _Float16* w1h  = hb + ho; ho += (size_t)DD * D2D;       // edge W1 (f16)
    _Float16* w2h  = hb + ho; ho += (size_t)(DD/2) * DD;    // edge W2 (f16)
    _Float16* elnh = hb + ho; ho += (size_t)CEC * D2D;      // edge LN rows
    _Float16* eh1h = hb + ho; ho += (size_t)CEC * DD;       // edge h1 (f16)
    if (ws_size < off * sizeof(float) + ho * sizeof(_Float16)) return;

    const int ND = NN * DD;
    auto cvt = [&](const float* s, int ld, _Float16* d, int rows, int cols) {
        int n = rows * cols;
        gt_cvt_f16<<<(n + 255) / 256, 256, 0, stream>>>(s, ld, d, rows, cols);
    };

    // x = node_emb (f32 + f16 copies)
    hipMemcpyAsync(xbuf, node_emb, (size_t)ND * sizeof(float),
                   hipMemcpyDeviceToDevice, stream);
    cvt(node_emb, DD, xh, NN, DD);

    // Degrees (with self loops): deg = 1 + count(dst); then deg^-1/2.
    gt_fill<<<NN / 256, 256, 0, stream>>>(degb, 1.0f, NN);
    gt_deg_accum<<<EE / 256, 256, 0, stream>>>(degb, dst, EE);
    gt_rsqrt_ip<<<NN / 256, 256, 0, stream>>>(degb, NN);

    // ---- GCN layers ----
    for (int l = 0; l < 2; ++l) {
        cvt(gcn[l].W, DD, whb, DD, DD);
        gemm16(stream, xh, DD, whb, DD, t1, nullptr, DD, nullptr, nullptr,
               NN, DD, DD, 0);                                   // xw = x@W^T
        gt_gcn_self<<<ND / 256, 256, 0, stream>>>(t1, degb, gcn[l].b, newx, ND);
        gt_gcn_edge<<<EE * (DD / 256), 256, 0, stream>>>(t1, src, dst, degb,
                                                         newx, EE * DD);
        // x = x + relu(LN(newx))  (f32 and f16 copies)
        gt_row_ln<<<NN, 256, 0, stream>>>(newx, xbuf, xbuf, xh, gcn[l].g,
                                          gcn[l].be, DD, 1 | 2);
    }

    // ---- Positional encoding ----
    gt_pe_add<<<ND / 256, 256, 0, stream>>>(xbuf, ND);

    // ---- Transformer layers ----
    for (int l = 0; l < 2; ++l) {
        gt_row_ln<<<NN, 256, 0, stream>>>(xbuf, nullptr, nullptr, lnh,
                                          tr[l].g1, tr[l].b1n, DD, 0);
        cvt(tr[l].Wqkv, DD, whb, 3 * DD, DD);
        gemm16(stream, lnh, DD, whb, DD, nullptr, qkvh, 3 * DD,
               tr[l].bqkv, nullptr, NN, 3 * DD, DD, 0);
        for (int h = 0; h < HH; ++h) {
            const _Float16* Qh = qkvh + h * DHD;           // [N,64] ld 1536
            const _Float16* Kh = qkvh + DD + h * DHD;
            const _Float16* Vh = qkvh + 2 * DD + h * DHD;  // [N,64] ld 1536
            gemm16(stream, Qh, 3 * DD, Kh, 3 * DD, nullptr, scbh, NN,
                   nullptr, nullptr, NN, NN, DHD, 0);      // scores
            gt_softmax_rows<<<NN, 256, 0, stream>>>(scbh, NN, 0.125f);
            gt_cvt_t_f16<<<(NN * DHD + 255) / 256, 256, 0, stream>>>(
                Vh, 3 * DD, vth, NN, DHD);                 // V^T [64, 4096]
            gemm16(stream, scbh, NN, vth, NN, obuf + h * DHD, nullptr, DD,
                   nullptr, nullptr, NN, DHD, NN, 0);      // attn @ V
        }
        cvt(obuf, DD, oh, NN, DD);
        cvt(tr[l].Wo, DD, whb, DD, DD);
        gemm16(stream, oh, DD, whb, DD, xbuf, nullptr, DD, tr[l].bo, xbuf,
               NN, DD, DD, 0);                             // x += o@Wo^T+bo
        gt_row_ln<<<NN, 256, 0, stream>>>(xbuf, nullptr, nullptr, lnh,
                                          tr[l].g2, tr[l].b2n, DD, 0);
        cvt(tr[l].W1, DD, whb, DFFN, DD);
        gemm16(stream, lnh, DD, whb, DD, nullptr, t1h, DFFN, tr[l].bf1,
               nullptr, NN, DFFN, DD, F_RELU);             // relu FFN hidden
        cvt(tr[l].W2, DFFN, whb, DD, DFFN);
        gemm16(stream, t1h, DFFN, whb, DFFN, xbuf, nullptr, DD, tr[l].bf2,
               xbuf, NN, DD, DFFN, 0);                     // x += f1@W2^T+bf2
    }

    // ---- Edge MLP (chunked) ----
    cvt(mW1, D2D, w1h, DD, D2D);
    cvt(mW2, DD, w2h, DD / 2, DD);
    for (int e0 = 0; e0 < EE; e0 += CEC) {
        gt_edge_gather_ln<<<CEC, 256, 0, stream>>>(xbuf, src + e0, dst + e0,
                                                   pre_g, pre_b, elnh);
        gemm16(stream, elnh, D2D, w1h, D2D, eh1, nullptr, DD, mb1, nullptr,
               CEC, DD, D2D, 0);
        gt_row_ln<<<CEC, 256, 0, stream>>>(eh1, nullptr, nullptr, eh1h,
                                           mg1, mbe1, DD, 1);
        gemm16(stream, eh1h, DD, w2h, DD, eh2, nullptr, DD / 2, mb2, nullptr,
               CEC, DD / 2, DD, 0);
        gt_row_ln<<<CEC, 256, 0, stream>>>(eh2, nullptr, eh2, nullptr,
                                           mg2, mbe2, DD / 2, 1);
        gt_edge_out<<<CEC / 256, 256, 0, stream>>>(eh2, mW3, mb3, outp + e0, CEC);
    }
}
